// Sparse_Transformer_76776835383952
// MI455X (gfx1250) — matrix operand order repaired and checked by reading, not compiled, not run on hardware
//
#include <hip/hip_runtime.h>
#include <stdint.h>

#define NN 16384
#define HH 4
#define CC 64
#define DD 256
#define FF 1024
#define LL 4

typedef __bf16 bf16_t;
typedef bf16_t v16bf __attribute__((ext_vector_type(16)));
typedef float  v8f   __attribute__((ext_vector_type(8)));

// ---------- helpers ----------
__device__ __forceinline__ bf16_t f2bf(float f) {
  unsigned u = __float_as_uint(f);
  unsigned r = (u + 0x7FFFu + ((u >> 16) & 1u)) >> 16;   // RNE fp32 -> bf16
  unsigned short h = (unsigned short)r;
  return __builtin_bit_cast(bf16_t, h);
}
__device__ __forceinline__ unsigned f_enc(float f) {     // monotone float->uint
  unsigned u = __float_as_uint(f);
  return (u & 0x80000000u) ? ~u : (u | 0x80000000u);
}
__device__ __forceinline__ float f_dec(unsigned u) {
  return __uint_as_float((u & 0x80000000u) ? (u ^ 0x80000000u) : ~u);
}

// ---------- elementwise: x = a (+ b), plus bf16 copy ----------
__global__ void addcvt_kernel(float* __restrict__ xf, bf16_t* __restrict__ xb,
                              const float* __restrict__ a, const float* __restrict__ b, int n) {
  int i = blockIdx.x * blockDim.x + threadIdx.x;
  if (i >= n) return;
  float v = a[i] + (b ? b[i] : 0.f);
  xf[i] = v;
  xb[i] = f2bf(v);
}

// ---------- weight convert + transpose: W[K,Ncol] f32 -> Wt[Ncol,K] bf16 ----------
__global__ void wconv_kernel(bf16_t* __restrict__ wt, const float* __restrict__ w,
                             int K, int Ncol) {
  int idx = blockIdx.x * blockDim.x + threadIdx.x;
  if (idx >= K * Ncol) return;
  int k = idx / Ncol, n = idx % Ncol;
  wt[(size_t)n * K + k] = f2bf(w[idx]);
}

// ---------- bf16 WMMA GEMM: out[M,Ncol] = A[M,K] @ Bt[Ncol,K]^T (+bias)(+ReLU) ----------
template <bool RELU>
__global__ __launch_bounds__(256)
void gemm_bf16_wmma(const bf16_t* __restrict__ A, const bf16_t* __restrict__ Bt,
                    const float* __restrict__ bias,
                    float* __restrict__ outF, bf16_t* __restrict__ outB,
                    int M, int K, int Ncol) {
  const int lane = threadIdx.x & 31;
  const int wave = threadIdx.x >> 5;
  const int wm = wave & 3, wn = wave >> 2;          // 4x2 waves -> 128x64 block tile
  const int mBase = blockIdx.y * 128 + wm * 32;
  const int nBase = blockIdx.x * 64 + wn * 32;
  const int l15  = lane & 15;
  const int half = lane >> 4;

  v8f z = {};
  v8f acc[2][2] = {{z, z}, {z, z}};

  union Frag { v16bf v; unsigned u[8]; };

  for (int k0 = 0; k0 < K; k0 += 32) {
    Frag a[2], b[2];
#pragma unroll
    for (int i = 0; i < 2; ++i) {                   // A frags: 16x32 bf16 layout
      const bf16_t* ap = A + (size_t)(mBase + i * 16 + l15) * K;
      if (k0 + 32 < K) __builtin_prefetch(ap + k0 + 32, 0, 1);
#pragma unroll
      for (int v = 0; v < 8; ++v) {
        int g = v >> 2, vv = v & 3;
        int k = k0 + g * 16 + half * 8 + vv * 2;    // packed K pair, contiguous
        a[i].u[v] = *reinterpret_cast<const unsigned*>(ap + k);
      }
    }
#pragma unroll
    for (int j = 0; j < 2; ++j) {                   // B frags from transposed weights
      const bf16_t* bp = Bt + (size_t)(nBase + j * 16 + l15) * K;
      int kb = k0 + half * 8;
#pragma unroll
      for (int v = 0; v < 8; ++v)
        b[j].u[v] = *reinterpret_cast<const unsigned*>(bp + kb + (v >> 2) * 16 + (v & 3) * 2);
    }
#pragma unroll
    for (int i = 0; i < 2; ++i)
#pragma unroll
      for (int j = 0; j < 2; ++j)
        acc[i][j] = __builtin_amdgcn_wmma_f32_16x16x32_bf16(
            false, a[i].v, false, b[j].v, (short)0, acc[i][j], false, false);
  }

#pragma unroll
  for (int i = 0; i < 2; ++i) {
#pragma unroll
    for (int j = 0; j < 2; ++j) {
      int col = nBase + j * 16 + l15;
      float bv = bias ? bias[col] : 0.f;
#pragma unroll
      for (int v = 0; v < 8; ++v) {
        int r = mBase + i * 16 + v + half * 8;
        float val = acc[i][j][v] + bv;
        if (RELU) val = val > 0.f ? val : 0.f;
        size_t off = (size_t)r * Ncol + col;
        if (outF) outF[off] = val;
        if (outB) outB[off] = f2bf(val);
      }
    }
  }
}

// ---------- attention coefficients: a_s[n,h], a_d[n,h] (float4 dot) ----------
__global__ void attn_coeff_kernel(float* __restrict__ as_, float* __restrict__ ad_,
                                  const float* __restrict__ h,
                                  const float* __restrict__ asr, const float* __restrict__ ads) {
  int t = blockIdx.x * blockDim.x + threadIdx.x;
  if (t >= NN * HH) return;
  int n = t / HH, hd = t % HH;
  const float4* hp = (const float4*)(h + (size_t)n * DD + hd * CC);
  const float4* sp = (const float4*)(asr + hd * CC);
  const float4* dp = (const float4*)(ads + hd * CC);
  float s0 = 0.f, s1 = 0.f;
#pragma unroll
  for (int c = 0; c < CC / 4; ++c) {
    float4 hv = hp[c], sv = sp[c], dv = dp[c];
    s0 += hv.x * sv.x + hv.y * sv.y + hv.z * sv.z + hv.w * sv.w;
    s1 += hv.x * dv.x + hv.y * dv.y + hv.z * dv.z + hv.w * dv.w;
  }
  as_[t] = s0;
  ad_[t] = s1;
}

// ---------- init segment buffers ----------
__global__ void seginit_kernel(unsigned* __restrict__ maxb, float* __restrict__ den,
                               float* __restrict__ agg) {
  int i = blockIdx.x * blockDim.x + threadIdx.x;
  if (i < NN * DD) agg[i] = 0.f;
  if (i < NN * HH) { maxb[i] = 0u; den[i] = 0.f; }
}

// ---------- edge pass 1: leaky-relu + segment max ----------
__global__ void edge_max_kernel(float* __restrict__ ebuf, unsigned* __restrict__ maxb,
                                const int* __restrict__ src, const int* __restrict__ dst,
                                const float* __restrict__ as_, const float* __restrict__ ad_,
                                int EH) {
  int t = blockIdx.x * blockDim.x + threadIdx.x;
  if (t >= EH) return;
  int e = t / HH, hd = t % HH;
  int s = src[e], d = dst[e];
  float v = as_[s * HH + hd] + ad_[d * HH + hd];
  v = v > 0.f ? v : 0.2f * v;                 // leaky_relu(0.2)
  ebuf[t] = v;
  atomicMax(&maxb[d * HH + hd], f_enc(v));
}

// ---------- edge pass 2: exp + segment sum ----------
__global__ void edge_exp_kernel(float* __restrict__ ebuf, float* __restrict__ den,
                                const unsigned* __restrict__ maxb,
                                const int* __restrict__ dst, int EH) {
  int t = blockIdx.x * blockDim.x + threadIdx.x;
  if (t >= EH) return;
  int e = t / HH, hd = t % HH;
  int d = dst[e];
  float ex = __expf(ebuf[t] - f_dec(maxb[d * HH + hd]));
  ebuf[t] = ex;
  atomicAdd(&den[d * HH + hd], ex);
}

// ---------- edge pass 3: normalize in place: ebuf -> alpha ----------
__global__ void edge_alpha_kernel(float* __restrict__ ebuf, const float* __restrict__ den,
                                  const int* __restrict__ dst, int EH) {
  int t = blockIdx.x * blockDim.x + threadIdx.x;
  if (t >= EH) return;
  int e = t / HH, hd = t % HH;
  ebuf[t] = ebuf[t] / fmaxf(den[dst[e] * HH + hd], 1e-16f);
}

// ---------- edge pass 4: weighted aggregation, one block per edge ----------
// 256 threads cover all D channels: coalesced h[src] gather and agg[dst] atomics.
__global__ __launch_bounds__(DD)
void edge_agg_kernel(float* __restrict__ agg, const float* __restrict__ alpha,
                     const float* __restrict__ h,
                     const int* __restrict__ src, const int* __restrict__ dst) {
  int e = blockIdx.x;
  int d = threadIdx.x;               // 0..255
  int hd = d >> 6;                   // d / CC
  float a = alpha[e * HH + hd];
  int s = src[e], ds = dst[e];
  atomicAdd(&agg[(size_t)ds * DD + d], h[(size_t)s * DD + d] * a);
}

// ---------- fused (acc + bias? + residual) -> LayerNorm -> f32 (+ bf16) ----------
__global__ __launch_bounds__(DD)
void ln_kernel(float* __restrict__ outF, bf16_t* __restrict__ outB,
               const float* __restrict__ acc, const float* __restrict__ bias,
               const float* __restrict__ resid,
               const float* __restrict__ g, const float* __restrict__ bt) {
  __shared__ float red[DD];
  int row = blockIdx.x, t = threadIdx.x;
  size_t off = (size_t)row * DD + t;
  float v = acc[off] + (bias ? bias[t] : 0.f) + resid[off];
  red[t] = v;
  __syncthreads();
  for (int s = DD / 2; s > 0; s >>= 1) {
    if (t < s) red[t] += red[t + s];
    __syncthreads();
  }
  float mean = red[0] * (1.f / DD);
  __syncthreads();
  float d = v - mean;
  red[t] = d * d;
  __syncthreads();
  for (int s = DD / 2; s > 0; s >>= 1) {
    if (t < s) red[t] += red[t + s];
    __syncthreads();
  }
  float var = red[0] * (1.f / DD);
  float y = d * rsqrtf(var + 1e-5f) * g[t] + bt[t];
  outF[off] = y;
  if (outB) outB[off] = f2bf(y);
}

// ---------------------------------------------------------------------------
extern "C" void kernel_launch(void* const* d_in, const int* in_sizes, int n_in,
                              void* d_out, int out_size, void* d_ws, size_t ws_size,
                              hipStream_t stream) {
  (void)n_in; (void)out_size; (void)ws_size;
  const float* in_hf = (const float*)d_in[0];
  const float* in_hs = (const float*)d_in[1];
  const int*   ei    = (const int*)d_in[2];
  const float* Wgat  = (const float*)d_in[3];
  const float* attS  = (const float*)d_in[4];
  const float* attD  = (const float*)d_in[5];
  const float* biasG = (const float*)d_in[6];
  const float* W1a   = (const float*)d_in[7];
  const float* b1a   = (const float*)d_in[8];
  const float* W2a   = (const float*)d_in[9];
  const float* b2a   = (const float*)d_in[10];
  const float* g1a   = (const float*)d_in[11];
  const float* bt1a  = (const float*)d_in[12];
  const float* g2a   = (const float*)d_in[13];
  const float* bt2a  = (const float*)d_in[14];

  const int E = in_sizes[2] / 2;
  const int* src = ei;
  const int* dst = ei + E;

  float* out_hf = (float*)d_out;
  float* out_hs = out_hf + (size_t)NN * DD;
  hipMemcpyAsync(out_hf, in_hf, sizeof(float) * (size_t)NN * DD, hipMemcpyDeviceToDevice, stream);
  hipMemcpyAsync(out_hs, in_hs, sizeof(float) * (size_t)NN * DD, hipMemcpyDeviceToDevice, stream);

  // workspace carve (256B aligned)
  char* p = (char*)d_ws;
  auto carve = [&](size_t bytes) -> void* {
    void* r = (void*)p;
    p += (bytes + 255) & ~(size_t)255;
    return r;
  };
  float*    xbuf = (float*)carve(sizeof(float) * (size_t)NN * DD);
  bf16_t*   xbf  = (bf16_t*)carve(sizeof(bf16_t) * (size_t)NN * DD);
  float*    hbuf = (float*)carve(sizeof(float) * (size_t)NN * DD);   // also reused as ybuf
  float*    asb  = (float*)carve(sizeof(float) * NN * HH);
  float*    adb  = (float*)carve(sizeof(float) * NN * HH);
  unsigned* maxb = (unsigned*)carve(sizeof(unsigned) * NN * HH);
  float*    denb = (float*)carve(sizeof(float) * NN * HH);
  float*    ebuf = (float*)carve(sizeof(float) * (size_t)E * HH);
  float*    agg  = (float*)carve(sizeof(float) * (size_t)NN * DD);
  float*    x1   = (float*)carve(sizeof(float) * (size_t)NN * DD);
  bf16_t*   x1bf = (bf16_t*)carve(sizeof(bf16_t) * (size_t)NN * DD);
  bf16_t*   fbuf = (bf16_t*)carve(sizeof(bf16_t) * (size_t)NN * FF);
  bf16_t*   wgt  = (bf16_t*)carve(sizeof(bf16_t) * DD * DD);
  bf16_t*   w1t  = (bf16_t*)carve(sizeof(bf16_t) * DD * FF);
  bf16_t*   w2t  = (bf16_t*)carve(sizeof(bf16_t) * FF * DD);

  const int TPB = 256;
  const int EH = E * HH;

  for (int l = 0; l < LL; ++l) {
    for (int s = 0; s < 2; ++s) {
      const int li = s * LL + l;
      const float* Wg  = Wgat + (size_t)li * DD * DD;
      const float* asr = attS + (size_t)li * HH * CC;
      const float* ads = attD + (size_t)li * HH * CC;
      const float* bg  = biasG + (size_t)li * DD;
      const float* W1  = W1a + (size_t)li * DD * FF;
      const float* b1  = b1a + (size_t)li * FF;
      const float* W2  = W2a + (size_t)li * FF * DD;
      const float* b2  = b2a + (size_t)li * DD;
      const float* g1  = g1a + (size_t)li * DD;
      const float* bt1 = bt1a + (size_t)li * DD;
      const float* g2  = g2a + (size_t)li * DD;
      const float* bt2 = bt2a + (size_t)li * DD;

      const float* xa = (s == 0) ? out_hf : out_hs;  // stream0: hf+hs, stream1: hs
      const float* xb = (s == 0) ? out_hs : nullptr;
      float* state_out = (s == 0) ? out_hf : out_hs;

      // x_in (f32 snapshot + bf16 operand)
      addcvt_kernel<<<(NN * DD + TPB - 1) / TPB, TPB, 0, stream>>>(xbuf, xbf, xa, xb, NN * DD);
      // h = x @ Wg
      wconv_kernel<<<(DD * DD + TPB - 1) / TPB, TPB, 0, stream>>>(wgt, Wg, DD, DD);
      gemm_bf16_wmma<false><<<dim3(DD / 64, NN / 128), TPB, 0, stream>>>(
          xbf, wgt, nullptr, hbuf, nullptr, NN, DD, DD);
      // attention coefficients + segment softmax + aggregation
      attn_coeff_kernel<<<(NN * HH + TPB - 1) / TPB, TPB, 0, stream>>>(asb, adb, hbuf, asr, ads);
      seginit_kernel<<<(NN * DD + TPB - 1) / TPB, TPB, 0, stream>>>(maxb, denb, agg);
      edge_max_kernel<<<(EH + TPB - 1) / TPB, TPB, 0, stream>>>(ebuf, maxb, src, dst, asb, adb, EH);
      edge_exp_kernel<<<(EH + TPB - 1) / TPB, TPB, 0, stream>>>(ebuf, denb, maxb, dst, EH);
      edge_alpha_kernel<<<(EH + TPB - 1) / TPB, TPB, 0, stream>>>(ebuf, denb, dst, EH);
      edge_agg_kernel<<<E, DD, 0, stream>>>(agg, ebuf, hbuf, src, dst);
      // x1 = LN(agg + bias_gat + x_in)
      ln_kernel<<<NN, DD, 0, stream>>>(x1, x1bf, agg, bg, xbuf, g1, bt1);
      // FFN: relu(x1@W1 + b1) @ W2 + b2
      wconv_kernel<<<(DD * FF + TPB - 1) / TPB, TPB, 0, stream>>>(w1t, W1, DD, FF);
      gemm_bf16_wmma<true><<<dim3(FF / 64, NN / 128), TPB, 0, stream>>>(
          x1bf, w1t, b1, nullptr, fbuf, NN, DD, FF);
      wconv_kernel<<<(FF * DD + TPB - 1) / TPB, TPB, 0, stream>>>(w2t, W2, FF, DD);
      gemm_bf16_wmma<false><<<dim3(DD / 64, NN / 128), TPB, 0, stream>>>(
          fbuf, w2t, b2, hbuf, nullptr, NN, FF, DD);
      // out = LN(ffn + x1), written in place into the stream state
      ln_kernel<<<NN, DD, 0, stream>>>(state_out, nullptr, hbuf, nullptr, x1, g2, bt2);
    }
  }
}
